// PEER_Adaptor_VP_90340342104132
// MI455X (gfx1250) — matrix operand order repaired and checked by reading, not compiled, not run on hardware
//
#include <hip/hip_runtime.h>
#include <math.h>

typedef __attribute__((ext_vector_type(16))) _Float16 v16h;
typedef __attribute__((ext_vector_type(8)))  _Float16 v8h;
typedef __attribute__((ext_vector_type(8)))  float    v8f;

// ---- model constants ----
#define B_    128
#define D_    768
#define L_    589           // 1 + 196*3
#define NK_   40
#define NTOK_ 196
#define ROWS1_ (B_*588)     // 75264 = 16*4704
#define ROWS2_ (B_*NTOK_)   // 25088 = 16*1568

__device__ __forceinline__ float qgelu_f(float v) {
  return v / (1.0f + __expf(-1.702f * v));
}
__device__ __forceinline__ float sigmoid_f(float v) {
  return 1.0f / (1.0f + __expf(-v));
}

// ---------------------------------------------------------------------------
// Async global->LDS copy (CDNA5 GLOBAL_LOAD_ASYNC_TO_LDS_B128, ASYNCcnt).
// Each active lane copies 16 bytes; per-wave ASYNCcnt +1 per instruction.
// ---------------------------------------------------------------------------
__device__ __forceinline__ void async_copy_b128(void* lds_dst, const void* gsrc) {
  unsigned off = (unsigned)(unsigned long long)lds_dst;   // LDS offset = addr[31:0]
  asm volatile("global_load_async_to_lds_b128 %0, %1, off"
               :: "v"(off), "v"(gsrc) : "memory");
}
#define WAIT_ASYNC(n) asm volatile("s_wait_asynccnt " #n ::: "memory")

// ---------------------------------------------------------------------------
// WMMA core: D(16x16,f32) += A(16xK) * W^T, A in LDS f16 row-major (ld = K),
// Wn = this lane's output-channel row of f16 weights (length K).
// Fragment layouts used here (wave32; contiguous k order, same for A and B):
//   A (16-bit 16x32): g=lane/16, m=lane%16; halves 0-7  = A[m][k0+16g..+7]
//                                           halves 8-15 = A[m][k0+16g+8..+7]
//   B (16-bit 32x16): half t = B[k0+16g+t][n]  (= W[n][k0+16g+t], contiguous)
// ---------------------------------------------------------------------------
__device__ __forceinline__ v8f wmma_acc_f16(const _Float16* As, int ldA,
                                            const _Float16* Wn, int Ksz, v8f acc) {
  const int lane = threadIdx.x & 31;
  const int g = lane >> 4;
  const int m = lane & 15;
  const _Float16* arow = As + m * ldA;
  for (int k0 = 0; k0 < Ksz; k0 += 32) {
    v8h alo = *(const v8h*)(arow + k0 + 16 * g);
    v8h ahi = *(const v8h*)(arow + k0 + 16 * g + 8);
    v16h a = __builtin_shufflevector(alo, ahi, 0,1,2,3,4,5,6,7,8,9,10,11,12,13,14,15);
    v16h b = *(const v16h*)(Wn + k0 + 16 * g);
    acc = __builtin_amdgcn_wmma_f32_16x16x32_f16(false, a, false, b, (short)0, acc,
                                                 false, false);
  }
  return acc;
}

// ---------------------------------------------------------------------------
__global__ void k_cvt_f16(const float* __restrict__ s, _Float16* __restrict__ d, int n) {
  int i = blockIdx.x * 256 + threadIdx.x;
  if (i < n) d[i] = (_Float16)s[i];
}

// Tap-decomposed conv weights: wt[tap][c][ci] f16, center tap gets -0.7*ksum fold.
__global__ void k_prep_tapw(const float* __restrict__ w, _Float16* __restrict__ wt,
                            int IC) {
  const int t = blockIdx.x * 256 + threadIdx.x;   // over 64*IC
  if (t >= 64 * IC) return;
  const int c = t / IC, ci = t % IC;
  const float* w9 = w + (size_t)(c * IC + ci) * 9;
  float ks = 0.f;
#pragma unroll
  for (int k = 0; k < 9; ++k) ks += w9[k];
#pragma unroll
  for (int k = 0; k < 9; ++k) {
    const float v = w9[k] - (k == 4 ? 0.7f * ks : 0.f);
    wt[((size_t)k * 64 + c) * IC + ci] = (_Float16)v;
  }
}

// K1: C = qgelu(x[:,1:589] @ c_down_w^T + b), rearranged to (B,196,192)
__global__ void k1_cdown(const float* __restrict__ x, const _Float16* __restrict__ Wh,
                         const float* __restrict__ bias, float* __restrict__ Cbuf) {
  __shared__ __align__(16) _Float16 As[16 * 768];
  const int rt = blockIdx.x;
  for (int t = threadIdx.x; t < 16 * 192; t += 128) {   // float4 granules
    int rloc = t / 192;
    int c4 = (t % 192) * 4;
    int row = rt * 16 + rloc;
    int b = row / 588, l = row % 588;
    const float4 v = *(const float4*)(x + ((size_t)(b * 589 + 1 + l)) * 768 + c4);
    _Float16* dst = As + rloc * 768 + c4;
    dst[0] = (_Float16)v.x; dst[1] = (_Float16)v.y;
    dst[2] = (_Float16)v.z; dst[3] = (_Float16)v.w;
  }
  __syncthreads();
  const int w = threadIdx.x >> 5;
  const int lane = threadIdx.x & 31;
  const int nn = lane & 15, g = lane >> 4;
  const int o = w * 16 + nn;
  v8f acc = {0.f,0.f,0.f,0.f,0.f,0.f,0.f,0.f};
  acc = wmma_acc_f16(As, 768, Wh + (size_t)o * 768, 768, acc);
  const float bo = bias[o];
  for (int r = 0; r < 8; ++r) {
    int row = rt * 16 + r + 8 * g;
    int b = row / 588, l = row % 588;
    int m = l / 196, n = l % 196;
    Cbuf[((size_t)(b * 196 + n)) * 192 + m * 64 + o] = qgelu_f(acc[r] + bo);
  }
}

// K2: G = concat(qgelu(C@g_down^T+b), qgelu(VP@vp_lin^T+b)) -> (B,196,128)
__global__ void k2_gdown_vplin(const float* __restrict__ Cbuf, const float* __restrict__ VP,
                               const _Float16* __restrict__ GWh, const float* __restrict__ gb,
                               const _Float16* __restrict__ VWh, const float* __restrict__ vb,
                               float* __restrict__ Gbuf) {
  __shared__ __align__(16) _Float16 A1[16 * 192];
  __shared__ __align__(16) _Float16 A2[16 * 64];
  const int rt = blockIdx.x;
  for (int t = threadIdx.x; t < 16 * 192; t += 256)
    A1[t] = (_Float16)Cbuf[(size_t)(rt * 16 + t / 192) * 192 + (t % 192)];
  for (int t = threadIdx.x; t < 16 * 64; t += 256)
    A2[t] = (_Float16)VP[(size_t)(rt * 16 + t / 64) * 64 + (t % 64)];
  __syncthreads();
  const int w = threadIdx.x >> 5;
  const int lane = threadIdx.x & 31;
  const int nn = lane & 15, g = lane >> 4;
  v8f acc = {0.f,0.f,0.f,0.f,0.f,0.f,0.f,0.f};
  if (w < 4) {
    const int o = w * 16 + nn;
    acc = wmma_acc_f16(A1, 192, GWh + (size_t)o * 192, 192, acc);
    const float bo = gb[o];
    for (int r = 0; r < 8; ++r) {
      int row = rt * 16 + r + 8 * g;
      Gbuf[(size_t)row * 128 + o] = qgelu_f(acc[r] + bo);
    }
  } else {
    const int o = (w - 4) * 16 + nn;
    acc = wmma_acc_f16(A2, 64, VWh + (size_t)o * 64, 64, acc);
    const float bo = vb[o];
    for (int r = 0; r < 8; ++r) {
      int row = rt * 16 + r + 8 * g;
      Gbuf[(size_t)row * 128 + 64 + o] = qgelu_f(acc[r] + bo);
    }
  }
}

// K3a: vp+gate conv_cd as 9 tap-GEMMs, WMMA, async double-buffered tap weights.
// Rows = (b,pos), pos = j*14+i; in(ci,j,i) = src[b][i*14+j][ci] (token-major).
// Output channels: waves 0-3 -> gate (ch 64..127), waves 4-7 -> vp (ch 0..63).
__global__ void k3a_wmma(const float* __restrict__ VP, const float* __restrict__ Gbuf,
                         const _Float16* __restrict__ WgT,   // (9,64,128)
                         const _Float16* __restrict__ WvT,   // (9,64,64)
                         float* __restrict__ vbuf) {
  __shared__ __align__(16) _Float16 Ag[16 * 128];
  __shared__ __align__(16) _Float16 Av[16 * 64];
  __shared__ __align__(16) _Float16 Wg[2][64 * 128];
  __shared__ __align__(16) _Float16 Wv[2][64 * 64];
  const int rt = blockIdx.x;
  const int tid = threadIdx.x;

  auto issue_w = [&](int tap, int buf) {   // 4+2 = 6 async instrs per wave
#pragma unroll
    for (int t = 0; t < 4; ++t) {
      const int idx = (t * 256 + tid) * 8;          // halves, 16B granules
      async_copy_b128(&Wg[buf][idx], WgT + (size_t)tap * 64 * 128 + idx);
    }
#pragma unroll
    for (int t = 0; t < 2; ++t) {
      const int idx = (t * 256 + tid) * 8;
      async_copy_b128(&Wv[buf][idx], WvT + (size_t)tap * 64 * 64 + idx);
    }
  };
  issue_w(0, 0);

  const int w = tid >> 5;
  const int lane = tid & 31;
  const int nn = lane & 15, g = lane >> 4;
  v8f acc = {0.f,0.f,0.f,0.f,0.f,0.f,0.f,0.f};

  for (int tap = 0; tap < 9; ++tap) {
    const int dj = tap / 3 - 1, di = tap % 3 - 1;
    for (int t = tid; t < 16 * 128; t += 256) {     // stage shifted gate A
      const int rloc = t >> 7, ci = t & 127;
      const int row = rt * 16 + rloc;
      const int b = row / 196, pos = row % 196;
      const int j = pos / 14, i = pos % 14;
      const int jj = j + dj, ii = i + di;
      float v = 0.f;
      if (jj >= 0 && jj < 14 && ii >= 0 && ii < 14)
        v = Gbuf[((size_t)b * 196 + (ii * 14 + jj)) * 128 + ci];
      Ag[t] = (_Float16)v;
    }
    for (int t = tid; t < 16 * 64; t += 256) {      // stage shifted vp A
      const int rloc = t >> 6, ci = t & 63;
      const int row = rt * 16 + rloc;
      const int b = row / 196, pos = row % 196;
      const int j = pos / 14, i = pos % 14;
      const int jj = j + dj, ii = i + di;
      float v = 0.f;
      if (jj >= 0 && jj < 14 && ii >= 0 && ii < 14)
        v = VP[((size_t)b * 196 + (ii * 14 + jj)) * 64 + ci];
      Av[t] = (_Float16)v;
    }
    if (tap < 8) { issue_w(tap + 1, (tap + 1) & 1); WAIT_ASYNC(6); }
    else         { WAIT_ASYNC(0); }
    __syncthreads();
    if (w < 4) acc = wmma_acc_f16(Ag, 128, &Wg[tap & 1][(w * 16 + nn) * 128], 128, acc);
    else       acc = wmma_acc_f16(Av, 64,  &Wv[tap & 1][((w - 4) * 16 + nn) * 64], 64, acc);
    __syncthreads();
  }
  const int ch = (w < 4) ? (64 + w * 16 + nn) : ((w - 4) * 16 + nn);
  for (int r = 0; r < 8; ++r) {
    const int row = rt * 16 + r + 8 * g;
    const int b = row / 196, pos = row % 196;
    vbuf[((size_t)b * 128 + ch) * 196 + pos] = qgelu_f(acc[r]);
  }
}

// K3b: channel means -> eca 3-tap conv -> sigmoid -> scale vbuf in place
__global__ void k3b_eca_scale(float* __restrict__ vbuf, const float* __restrict__ eca) {
  __shared__ float y[130];
  const int b = blockIdx.x;
  const int c = threadIdx.x;      // 128 threads
  float* vrow = vbuf + ((size_t)b * 128 + c) * 196;
  float s = 0.f;
  for (int p = 0; p < 196; ++p) s += vrow[p];
  y[c + 1] = s * (1.0f / 196.0f);
  if (c == 0) { y[0] = 0.f; y[129] = 0.f; }
  __syncthreads();
  const float y2 = eca[0] * y[c] + eca[1] * y[c + 1] + eca[2] * y[c + 2];
  const float sg = sigmoid_f(y2);
  for (int p = 0; p < 196; ++p) vrow[p] *= sg;
}

// K3c: conv_comp as 9 tap-GEMMs on scaled v; input channel-major (B,128,196).
__global__ void k3c_wmma(const float* __restrict__ vbufS, const _Float16* __restrict__ WcT,
                         float* __restrict__ outVP) {
  __shared__ __align__(16) _Float16 Ac[16 * 128];
  __shared__ __align__(16) _Float16 Wc[2][64 * 128];
  const int rt = blockIdx.x;
  const int tid = threadIdx.x;   // 128 threads, 4 waves

  auto issue_w = [&](int tap, int buf) {   // 8 async instrs per wave
#pragma unroll
    for (int t = 0; t < 8; ++t) {
      const int idx = (t * 128 + tid) * 8;
      async_copy_b128(&Wc[buf][idx], WcT + (size_t)tap * 64 * 128 + idx);
    }
  };
  issue_w(0, 0);

  const int w = tid >> 5;
  const int lane = tid & 31;
  const int nn = lane & 15, g = lane >> 4;
  v8f acc = {0.f,0.f,0.f,0.f,0.f,0.f,0.f,0.f};

  for (int tap = 0; tap < 9; ++tap) {
    const int dj = tap / 3 - 1, di = tap % 3 - 1;
    for (int t = tid; t < 16 * 128; t += 128) {
      const int rloc = t >> 7, ci = t & 127;
      const int row = rt * 16 + rloc;
      const int b = row / 196, pos = row % 196;
      const int j = pos / 14, i = pos % 14;
      const int jj = j + dj, ii = i + di;
      float v = 0.f;
      if (jj >= 0 && jj < 14 && ii >= 0 && ii < 14)
        v = vbufS[((size_t)b * 128 + ci) * 196 + jj * 14 + ii];
      Ac[t] = (_Float16)v;
    }
    if (tap < 8) { issue_w(tap + 1, (tap + 1) & 1); WAIT_ASYNC(8); }
    else         { WAIT_ASYNC(0); }
    __syncthreads();
    acc = wmma_acc_f16(Ac, 128, &Wc[tap & 1][(w * 16 + nn) * 128], 128, acc);
    __syncthreads();
  }
  const int c = w * 16 + nn;
  for (int r = 0; r < 8; ++r) {
    const int row = rt * 16 + r + 8 * g;
    const int b = row / 196, pos = row % 196;
    outVP[((size_t)b * 196 + pos) * 64 + c] = qgelu_f(acc[r]);
  }
}

// K4: rmsnorm of Cbuf (192, gamma_norm) and Gbuf (128, gamma_gate); 1 wave/row
__global__ void k4_rms(const float* __restrict__ Cbuf, const float* __restrict__ Gbuf,
                       const float* __restrict__ gnorm, const float* __restrict__ ggate,
                       float* __restrict__ xnbuf, float* __restrict__ gnbuf) {
  const int wave = blockIdx.x * 8 + (threadIdx.x >> 5);
  const int lane = threadIdx.x & 31;
  const float* src; float* dst; const float* gamma; int len; float sc;
  if (wave < ROWS2_) {
    src = Cbuf + (size_t)wave * 192; dst = xnbuf + (size_t)wave * 192;
    gamma = gnorm; len = 192; sc = 13.856406460551018f;      // sqrt(192)
  } else {
    const int r = wave - ROWS2_;
    src = Gbuf + (size_t)r * 128; dst = gnbuf + (size_t)r * 128;
    gamma = ggate; len = 128; sc = 11.313708498984761f;      // sqrt(128)
  }
  float ss = 0.f;
  for (int i = lane; i < len; i += 32) { const float v = src[i]; ss += v * v; }
  for (int off = 16; off; off >>= 1) ss += __shfl_xor(ss, off, 32);
  const float n = sqrtf(ss);
  const float k = sc / fmaxf(n, 1e-12f);
  for (int i = lane; i < len; i += 32) dst[i] = src[i] * k * gamma[i];
}

// K5a: q = gn @ to_q_w^T -> qbuf (ROWS2, 256)
__global__ void k5a_q(const float* __restrict__ gnbuf, const _Float16* __restrict__ QWh,
                      float* __restrict__ qbuf) {
  __shared__ __align__(16) _Float16 As[16 * 128];
  const int rt = blockIdx.x;
  for (int t = threadIdx.x; t < 16 * 128; t += 512)
    As[t] = (_Float16)gnbuf[(size_t)(rt * 16 + t / 128) * 128 + (t % 128)];
  __syncthreads();
  const int w = threadIdx.x >> 5;
  const int lane = threadIdx.x & 31;
  const int nn = lane & 15, g = lane >> 4;
  const int o = w * 16 + nn;
  v8f acc = {0.f,0.f,0.f,0.f,0.f,0.f,0.f,0.f};
  acc = wmma_acc_f16(As, 128, QWh + (size_t)o * 128, 128, acc);
  for (int r = 0; r < 8; ++r) {
    const int row = rt * 16 + r + 8 * g;
    qbuf[(size_t)row * 256 + o] = acc[r];
  }
}

// K5b: sims vs keys, product-key top-k (K=2 each axis, then top-2 of 4), softmax
__global__ void k5b_topk(const float* __restrict__ qbuf, const float* __restrict__ keys,
                         int* __restrict__ idxb, float* __restrict__ coefb) {
  const int gid = blockIdx.x * 256 + threadIdx.x;
  if (gid >= ROWS2_ * 2) return;
  const int bn = gid >> 1, h = gid & 1;
  const float* q0 = qbuf + (size_t)bn * 256 + h * 64;          // p=0
  const float* q1 = qbuf + (size_t)bn * 256 + 128 + h * 64;    // p=1
  float sx0 = -INFINITY, sx1 = -INFINITY; int ix0 = 0, ix1 = 0;
  float sy0 = -INFINITY, sy1 = -INFINITY; int iy0 = 0, iy1 = 0;
  for (int k = 0; k < NK_; ++k) {
    const float* kp0 = keys + (((size_t)h * NK_ + k) * 2 + 0) * 64;
    const float* kp1 = keys + (((size_t)h * NK_ + k) * 2 + 1) * 64;
    float d0 = 0.f, d1 = 0.f;
    for (int d = 0; d < 64; ++d) { d0 += q0[d] * kp0[d]; d1 += q1[d] * kp1[d]; }
    if (d0 > sx0)      { sx1 = sx0; ix1 = ix0; sx0 = d0; ix0 = k; }
    else if (d0 > sx1) { sx1 = d0; ix1 = k; }
    if (d1 > sy0)      { sy1 = sy0; iy1 = iy0; sy0 = d1; iy0 = k; }
    else if (d1 > sy1) { sy1 = d1; iy1 = k; }
  }
  float cs[4]; int ci[4];
  cs[0] = sx0 + sy0; ci[0] = ix0 * NK_ + iy0;
  cs[1] = sx0 + sy1; ci[1] = ix0 * NK_ + iy1;
  cs[2] = sx1 + sy0; ci[2] = ix1 * NK_ + iy0;
  cs[3] = sx1 + sy1; ci[3] = ix1 * NK_ + iy1;
  float b0 = -INFINITY, b1 = -INFINITY; int e0 = 0, e1 = 0;
  for (int t = 0; t < 4; ++t) {
    if (cs[t] > b0)      { b1 = b0; e1 = e0; b0 = cs[t]; e0 = ci[t]; }
    else if (cs[t] > b1) { b1 = cs[t]; e1 = ci[t]; }
  }
  const float w1 = __expf(b1 - b0);
  const float inv = 1.f / (1.f + w1);
  const size_t base = ((size_t)bn * 2 + h) * 2;
  idxb[base + 0] = e0;  idxb[base + 1] = e1;
  coefb[base + 0] = inv; coefb[base + 1] = w1 * inv;
}

// K6: Cp[b,n,:] = sum_{h,k} gelu(xn . w_down[e]) * softmax * w_up[e]; 1 wave/token
__global__ void k6_expert(const float* __restrict__ xnbuf, const int* __restrict__ idxb,
                          const float* __restrict__ coefb, const float* __restrict__ wdown,
                          const float* __restrict__ wup, float* __restrict__ Cpbuf) {
  const int wave = blockIdx.x * 8 + (threadIdx.x >> 5);
  const int lane = threadIdx.x & 31;
  const float* xr = xnbuf + (size_t)wave * 192;
  float xv[6], acc[6];
#pragma unroll
  for (int j = 0; j < 6; ++j) { xv[j] = xr[lane + 32 * j]; acc[j] = 0.f; }
  for (int hk = 0; hk < 4; ++hk) {
    const int e = idxb[(size_t)wave * 4 + hk];
    const float cw = coefb[(size_t)wave * 4 + hk];
    const float* wd = wdown + (size_t)e * 192;
    const float* wu = wup + (size_t)e * 192;
    float p = 0.f;
#pragma unroll
    for (int j = 0; j < 6; ++j) p += xv[j] * wd[lane + 32 * j];
    for (int off = 16; off; off >>= 1) p += __shfl_xor(p, off, 32);
    const float gl = 0.5f * p * (1.f + erff(p * 0.7071067811865476f));
    const float coef = gl * cw;
#pragma unroll
    for (int j = 0; j < 6; ++j) acc[j] += coef * wu[lane + 32 * j];
  }
  float* cp = Cpbuf + (size_t)wave * 192;
#pragma unroll
  for (int j = 0; j < 6; ++j) cp[lane + 32 * j] = acc[j];
}

// K7: out1[b,1+l,:] = x + qgelu(Cp2 @ conv_up_w^T + b); Cp2[b,m*196+n,o]=Cp[b,n,m*64+o]
__global__ void k7_convup(const float* __restrict__ Cpbuf, const _Float16* __restrict__ UWh,
                          const float* __restrict__ ub, const float* __restrict__ x,
                          float* __restrict__ out1) {
  __shared__ __align__(16) _Float16 As[16 * 64];
  const int rt = blockIdx.x;
  for (int t = threadIdx.x; t < 16 * 64; t += 128) {
    const int rloc = t / 64, o = t % 64;
    const int row = rt * 16 + rloc;
    const int b = row / 588, l = row % 588;
    const int m = l / 196, n = l % 196;
    As[t] = (_Float16)Cpbuf[((size_t)(b * 196 + n)) * 192 + m * 64 + o];
  }
  __syncthreads();
  const int w = threadIdx.x >> 5;
  const int lane = threadIdx.x & 31;
  const int nn = lane & 15, g = lane >> 4;
  const int d = (blockIdx.y * 4 + w) * 16 + nn;
  v8f acc = {0.f,0.f,0.f,0.f,0.f,0.f,0.f,0.f};
  acc = wmma_acc_f16(As, 64, UWh + (size_t)d * 64, 64, acc);
  const float bd = ub[d];
  for (int r = 0; r < 8; ++r) {
    const int row = rt * 16 + r + 8 * g;
    const int b = row / 588, l = row % 588;
    const size_t off = ((size_t)(b * 589) + 1 + l) * 768 + d;
    out1[off] = x[off] + qgelu_f(acc[r] + bd);
  }
}

// K8: cls row -> 2*x
__global__ void k8_cls(const float* __restrict__ x, float* __restrict__ out1) {
  const int t = blockIdx.x * 256 + threadIdx.x;    // 128*768
  const int b = t / 768, d = t % 768;
  const size_t off = (size_t)b * 589 * 768 + d;
  out1[off] = 2.f * x[off];
}

// ---------------------------------------------------------------------------
extern "C" void kernel_launch(void* const* d_in, const int* in_sizes, int n_in,
                              void* d_out, int out_size, void* d_ws, size_t ws_size,
                              hipStream_t stream) {
  (void)in_sizes; (void)n_in; (void)out_size; (void)ws_size;
  const float* x          = (const float*)d_in[0];
  const float* VP         = (const float*)d_in[1];
  const float* c_down_w   = (const float*)d_in[2];
  const float* c_down_b   = (const float*)d_in[3];
  const float* g_down_w   = (const float*)d_in[4];
  const float* g_down_b   = (const float*)d_in[5];
  const float* vp_lin_w   = (const float*)d_in[6];
  const float* vp_lin_b   = (const float*)d_in[7];
  const float* conv_up_w  = (const float*)d_in[8];
  const float* conv_up_b  = (const float*)d_in[9];
  const float* gamma_norm = (const float*)d_in[10];
  const float* gamma_gate = (const float*)d_in[11];
  const float* to_q_w     = (const float*)d_in[12];
  const float* keys       = (const float*)d_in[13];
  const float* w_down     = (const float*)d_in[14];
  const float* w_up       = (const float*)d_in[15];
  const float* conv_vp_k  = (const float*)d_in[16];
  const float* conv_gate_k= (const float*)d_in[17];
  const float* conv_comp_k= (const float*)d_in[18];
  const float* eca_w      = (const float*)d_in[19];

  float* out1 = (float*)d_out;
  float* out2 = (float*)d_out + (size_t)B_ * L_ * D_;

  char* wsb = (char*)d_ws;
  size_t off = 0;
  auto alloc_f = [&](size_t n) {
    float* p = (float*)(wsb + off);
    off += n * sizeof(float);
    off = (off + 255) & ~(size_t)255;
    return p;
  };
  float* Cbuf  = alloc_f((size_t)ROWS2_ * 192);
  float* Gbuf  = alloc_f((size_t)ROWS2_ * 128);
  float* xnbuf = alloc_f((size_t)ROWS2_ * 192);
  float* gnbuf = alloc_f((size_t)ROWS2_ * 128);
  float* qbuf  = alloc_f((size_t)ROWS2_ * 256);
  float* Cpbuf = alloc_f((size_t)ROWS2_ * 192);
  float* vbuf  = alloc_f((size_t)B_ * 128 * 196);
  int*   idxb  = (int*)alloc_f((size_t)ROWS2_ * 4);
  float* coefb = alloc_f((size_t)ROWS2_ * 4);
  _Float16* cwh = (_Float16*)alloc_f(64 * 768 / 2);
  _Float16* gwh = (_Float16*)alloc_f(64 * 192 / 2);
  _Float16* vwh = (_Float16*)alloc_f(64 * 64 / 2);
  _Float16* qwh = (_Float16*)alloc_f(256 * 128 / 2);
  _Float16* uwh = (_Float16*)alloc_f(768 * 64 / 2);
  _Float16* wvT = (_Float16*)alloc_f(9 * 64 * 64 / 2);    // tap-adjusted conv weights
  _Float16* wgT = (_Float16*)alloc_f(9 * 64 * 128 / 2);
  _Float16* wcT = (_Float16*)alloc_f(9 * 64 * 128 / 2);

  auto cdiv = [](int a, int b) { return (a + b - 1) / b; };
  k_cvt_f16<<<cdiv(64 * 768, 256), 256, 0, stream>>>(c_down_w, cwh, 64 * 768);
  k_cvt_f16<<<cdiv(64 * 192, 256), 256, 0, stream>>>(g_down_w, gwh, 64 * 192);
  k_cvt_f16<<<cdiv(64 * 64, 256), 256, 0, stream>>>(vp_lin_w, vwh, 64 * 64);
  k_cvt_f16<<<cdiv(256 * 128, 256), 256, 0, stream>>>(to_q_w, qwh, 256 * 128);
  k_cvt_f16<<<cdiv(768 * 64, 256), 256, 0, stream>>>(conv_up_w, uwh, 768 * 64);
  k_prep_tapw<<<cdiv(64 * 64, 256), 256, 0, stream>>>(conv_vp_k, wvT, 64);
  k_prep_tapw<<<cdiv(64 * 128, 256), 256, 0, stream>>>(conv_gate_k, wgT, 128);
  k_prep_tapw<<<cdiv(64 * 128, 256), 256, 0, stream>>>(conv_comp_k, wcT, 128);

  k1_cdown<<<ROWS1_ / 16, 128, 0, stream>>>(x, cwh, c_down_b, Cbuf);
  k2_gdown_vplin<<<ROWS2_ / 16, 256, 0, stream>>>(Cbuf, VP, gwh, g_down_b, vwh,
                                                  vp_lin_b, Gbuf);
  k3a_wmma<<<ROWS2_ / 16, 256, 0, stream>>>(VP, Gbuf, wgT, wvT, vbuf);
  k3b_eca_scale<<<B_, 128, 0, stream>>>(vbuf, eca_w);
  k3c_wmma<<<ROWS2_ / 16, 128, 0, stream>>>(vbuf, wcT, out2);
  k4_rms<<<(2 * ROWS2_) / 8, 256, 0, stream>>>(Cbuf, Gbuf, gamma_norm, gamma_gate,
                                               xnbuf, gnbuf);
  k5a_q<<<ROWS2_ / 16, 512, 0, stream>>>(gnbuf, qwh, qbuf);
  k5b_topk<<<cdiv(ROWS2_ * 2, 256), 256, 0, stream>>>(qbuf, keys, idxb, coefb);
  k6_expert<<<ROWS2_ / 8, 256, 0, stream>>>(xnbuf, idxb, coefb, w_down, w_up, Cpbuf);
  k7_convup<<<dim3(ROWS1_ / 16, 12), 128, 0, stream>>>(Cpbuf, uwh, conv_up_b, x, out1);
  k8_cls<<<cdiv(B_ * D_, 256), 256, 0, stream>>>(x, out1);
}
